// SimpleMamba_37005438222490
// MI455X (gfx1250) — matrix operand order repaired and checked by reading, not compiled, not run on hardware
//
#include <hip/hip_runtime.h>

// ---------------------------------------------------------------------------
// Problem constants (from reference)
// ---------------------------------------------------------------------------
constexpr int Vc = 32000, Hc = 512, DEPTHc = 6;
constexpr int Bc = 8, Lc = 256;
constexpr int ADc = 1024, TDc = 1024;
constexpr int TAc = 200, TTc = 100;
constexpr int DIc = 1024, DSc = 16, DCc = 4, DTRc = 32;
constexpr int Mrows = Bc * Lc;              // 2048 token rows

// ---------------------------------------------------------------------------
// bf16 helpers (bit-level RNE conversion; storage type __bf16 for WMMA)
// ---------------------------------------------------------------------------
typedef __bf16 bf16_t;
typedef __attribute__((ext_vector_type(16))) __bf16 v16bf;
typedef __attribute__((ext_vector_type(8)))  float  v8f;

__device__ __forceinline__ unsigned short f2bf_u(float f) {
  unsigned u = __float_as_uint(f);
  u += 0x7FFFu + ((u >> 16) & 1u);          // round-to-nearest-even
  return (unsigned short)(u >> 16);
}
__device__ __forceinline__ bf16_t f2bf(float f) {
  unsigned short s = f2bf_u(f);
  bf16_t b; __builtin_memcpy(&b, &s, 2); return b;
}

// 4-wide vectorized f32 -> bf16 converter
__global__ __launch_bounds__(256) void sm_cvt_f32_bf16_x4(bf16_t* __restrict__ dst,
                                                          const float* __restrict__ src,
                                                          int n4) {
  int i = blockIdx.x * 256 + threadIdx.x;
  if (i >= n4) return;
  const float4 v = ((const float4*)src)[i];
  unsigned long long p = (unsigned long long)f2bf_u(v.x)
                       | ((unsigned long long)f2bf_u(v.y) << 16)
                       | ((unsigned long long)f2bf_u(v.z) << 32)
                       | ((unsigned long long)f2bf_u(v.w) << 48);
  ((unsigned long long*)dst)[i] = p;
}

// dt_w [1024,32] f32 -> [1024,64] bf16, upper 32 columns zeroed (K padding so
// the K=32 GEMM can run through the K%64==0 WMMA pipeline).
__global__ __launch_bounds__(256) void sm_cvt_dtw_pad(bf16_t* __restrict__ dst,
                                                      const float* __restrict__ src) {
  int idx = blockIdx.x * 256 + threadIdx.x;         // DIc*64 elements
  if (idx >= DIc * 64) return;
  int row = idx >> 6, k = idx & 63;
  dst[idx] = (k < DTRc) ? f2bf(src[row * DTRc + k]) : f2bf(0.0f);
}

// ---------------------------------------------------------------------------
// Async global->LDS copy of 64 bytes (four b128 DMA transfers), tracked by
// ASYNCcnt; the DMA writes LDS directly (no VGPR round-trip).
// INST_OFFSET is added to both the LDS and global addresses (ISA ch8 §4.4).
// ---------------------------------------------------------------------------
__device__ __forceinline__ void sm_async_copy64B(const bf16_t* lds_dst,
                                                 const bf16_t* gsrc) {
  unsigned l = (unsigned)(uintptr_t)lds_dst;        // flat addr[31:0] == LDS offset
  unsigned long long g = (unsigned long long)(uintptr_t)gsrc;
  asm volatile("global_load_async_to_lds_b128 %0, %1, off\n\t"
               "global_load_async_to_lds_b128 %0, %1, off offset:16\n\t"
               "global_load_async_to_lds_b128 %0, %1, off offset:32\n\t"
               "global_load_async_to_lds_b128 %0, %1, off offset:48"
               :: "v"(l), "v"(g) : "memory");
}
__device__ __forceinline__ void sm_wait_async0() {
  asm volatile("s_wait_asynccnt 0x0" ::: "memory");
}

// ---------------------------------------------------------------------------
// Tiled bf16 WMMA GEMM:  C[M,N] (f32) = A[M,K](bf16,lda) * B[N,K](bf16,ldb)^T
// Optional bias[N]. Block tile 128x128x64, 8 waves (wave32); per K-slab each
// wave runs 2 half-slabs of a 2x4 grid of v_wmma_f32_16x16x32_bf16 (16 WMMAs
// per barrier pair). Double-buffered LDS (64KB) fed by async DMA.
// Requirements (all call sites satisfy): K % 64 == 0, 16B-aligned rows.
// ---------------------------------------------------------------------------
#define GBM 128
#define GBN 128
#define GBK 64

__global__ __launch_bounds__(256) void sm_gemm_bf16_wmma(
    float* __restrict__ C, const bf16_t* __restrict__ A,
    const bf16_t* __restrict__ Bw, const float* __restrict__ bias,
    int M, int N, int K, int lda, int ldb, int ldc) {
  __shared__ __align__(16) bf16_t As[2][GBM * GBK];
  __shared__ __align__(16) bf16_t Bs[2][GBN * GBK];

  const int tid  = threadIdx.x;
  const int wave = tid >> 5;
  const int lane = tid & 31;
  const int m16  = lane & 15;     // row (A) / col (B,C) within 16-tile
  const int hh   = lane >> 4;     // lane-half selector
  const int wm   = wave & 3;      // 4 row groups of 32
  const int wn   = wave >> 2;     // 2 col groups of 64
  const int blockM = blockIdx.y * GBM;
  const int blockN = blockIdx.x * GBN;

  v8f acc[2][4];
  v8f vz = {};
#pragma unroll
  for (int i = 0; i < 2; ++i)
#pragma unroll
    for (int j = 0; j < 4; ++j) acc[i][j] = vz;

  // staging: each thread owns a 64B slice (32 bf16) of one A row and one B row
  const int lrow = tid >> 1;            // 0..127
  const int lcol = (tid & 1) * 32;      // 0 or 32
  const int gr   = blockM + lrow;       // A row
  const int gn   = blockN + lrow;       // B (weight) row
  const bool aRow = (gr < M);
  const bool bRow = (gn < N);

  // rows that will never be DMA'd are zeroed once (both buffers)
  if (!aRow) {
    const bf16_t bz = f2bf(0.0f);
#pragma unroll
    for (int c = 0; c < 32; ++c) {
      As[0][lrow * GBK + lcol + c] = bz;
      As[1][lrow * GBK + lcol + c] = bz;
    }
  }
  if (!bRow) {
    const bf16_t bz = f2bf(0.0f);
#pragma unroll
    for (int c = 0; c < 32; ++c) {
      Bs[0][lrow * GBK + lcol + c] = bz;
      Bs[1][lrow * GBK + lcol + c] = bz;
    }
  }

  // issue async DMA for one K-slab into buffer `buf`
  auto issue = [&](int buf, int k0) {
    if (aRow) sm_async_copy64B(&As[buf][lrow * GBK + lcol],
                               &A[(size_t)gr * lda + k0 + lcol]);
    if (bRow) sm_async_copy64B(&Bs[buf][lrow * GBK + lcol],
                               &Bw[(size_t)gn * ldb + k0 + lcol]);
  };

  issue(0, 0);
  int cur = 0;

  for (int k0 = 0; k0 < K; k0 += GBK) {
    sm_wait_async0();        // this wave's DMA into buf[cur] done
    __syncthreads();         // everyone's DMA done -> buf[cur] readable

    if (k0 + GBK < K) issue(cur ^ 1, k0 + GBK);   // overlap next slab

    const bf16_t* Ab = As[cur];
    const bf16_t* Bb = Bs[cur];
    // two 32-wide K-halves, fragment registers reused across halves
#pragma unroll
    for (int half = 0; half < 2; ++half) {
      const int kk = half * 32;
      // ---- fragments per ISA 16-bit A(16x32)/B(32x16) layouts ----
      v16bf afrag[2];
#pragma unroll
      for (int i = 0; i < 2; ++i) {
        const bf16_t* ar = &Ab[(wm * 32 + i * 16 + m16) * GBK + kk];
#pragma unroll
        for (int j = 0; j < 8; ++j) {
          afrag[i][j]     = ar[8 * hh + j];        // K = 8h + j
          afrag[i][8 + j] = ar[16 + 8 * hh + j];   // K = 16 + 8h + j
        }
      }
      v16bf bfrag[4];
#pragma unroll
      for (int j = 0; j < 4; ++j) {
        const bf16_t* br = &Bb[(wn * 64 + j * 16 + m16) * GBK + kk];
#pragma unroll
        for (int q = 0; q < 8; ++q) {
          bfrag[j][q]     = br[8 * hh + q];        // K = 8h + q
          bfrag[j][8 + q] = br[16 + 8 * hh + q];   // K = 16 + 8h + q
        }
      }
#pragma unroll
      for (int i = 0; i < 2; ++i)
#pragma unroll
        for (int j = 0; j < 4; ++j)
          acc[i][j] = __builtin_amdgcn_wmma_f32_16x16x32_bf16(
              false, afrag[i], false, bfrag[j], (short)0, acc[i][j], false, false);
    }

    __syncthreads();         // all reads of buf[cur] done before it is re-filled
    cur ^= 1;
  }

  // ---- store D (C/D layout: row = 8h + r, col = lane%16) ----
  const bool fullTile = (blockM + GBM <= M) && (blockN + GBN <= N);
  if (fullTile) {
#pragma unroll
    for (int i = 0; i < 2; ++i) {
      const int rowBase = blockM + wm * 32 + i * 16 + 8 * hh;
#pragma unroll
      for (int j = 0; j < 4; ++j) {
        const int col = blockN + wn * 64 + j * 16 + m16;
        const float bv = bias ? bias[col] : 0.0f;
#pragma unroll
        for (int r = 0; r < 8; ++r)
          C[(size_t)(rowBase + r) * ldc + col] = acc[i][j][r] + bv;
      }
    }
  } else {
#pragma unroll
    for (int i = 0; i < 2; ++i) {
      const int rowBase = blockM + wm * 32 + i * 16 + 8 * hh;
#pragma unroll
      for (int j = 0; j < 4; ++j) {
        const int col = blockN + wn * 64 + j * 16 + m16;
        if (col < N) {
          const float bv = bias ? bias[col] : 0.0f;
#pragma unroll
          for (int r = 0; r < 8; ++r) {
            const int rr = rowBase + r;
            if (rr < M) C[(size_t)rr * ldc + col] = acc[i][j][r] + bv;
          }
        }
      }
    }
  }
}

// ---------------------------------------------------------------------------
// Embedding: x = tok_emb[ids] + pos_emb
// ---------------------------------------------------------------------------
__global__ __launch_bounds__(256) void sm_embed(float* __restrict__ x,
                                                const int* __restrict__ ids,
                                                const float* __restrict__ tok,
                                                const float* __restrict__ pos) {
  int idx = blockIdx.x * 256 + threadIdx.x;
  if (idx >= Mrows * Hc) return;
  int h  = idx & (Hc - 1);
  int bl = idx >> 9;            // / 512
  int l  = bl & (Lc - 1);
  x[idx] = tok[(size_t)ids[bl] * Hc + h] + pos[l * Hc + h];
}

// ---------------------------------------------------------------------------
// Masked sum over time for pooled conditioning
// ---------------------------------------------------------------------------
__global__ __launch_bounds__(256) void sm_mem_reduce(const float* __restrict__ feats,
                                                     const unsigned char* __restrict__ mask,
                                                     int T, int D,
                                                     float* __restrict__ sums,
                                                     float* __restrict__ cnt) {
  int b = blockIdx.x, tid = threadIdx.x;
  float acc0 = 0.f, acc1 = 0.f, acc2 = 0.f, acc3 = 0.f;
  int c = 0;
  for (int t = 0; t < T; ++t) {
    if (!mask[b * T + t]) {
      ++c;
      const float* row = feats + ((size_t)b * T + t) * D;
      acc0 += row[tid];       acc1 += row[tid + 256];
      acc2 += row[tid + 512]; acc3 += row[tid + 768];
    }
  }
  sums[b * D + tid]       = acc0; sums[b * D + tid + 256] = acc1;
  sums[b * D + tid + 512] = acc2; sums[b * D + tid + 768] = acc3;
  if (tid == 0) cnt[b] = (float)c;
}

__global__ __launch_bounds__(512) void sm_pooled(const float* __restrict__ sa,
                                                 const float* __restrict__ st,
                                                 const float* __restrict__ cnts,
                                                 const float* __restrict__ aw,
                                                 const float* __restrict__ ab,
                                                 const float* __restrict__ tw,
                                                 const float* __restrict__ tb,
                                                 const float* __restrict__ mod,
                                                 float* __restrict__ pooled) {
  int b = blockIdx.x, h = threadIdx.x;
  float na = cnts[b], nt = cnts[Bc + b];
  const float* sab = sa + b * ADc;
  const float* stb = st + b * TDc;
  float da = 0.f, dtv = 0.f;
  for (int k = 0; k < ADc; ++k) da  += sab[k] * aw[(size_t)h * ADc + k];
  for (int k = 0; k < TDc; ++k) dtv += stb[k] * tw[(size_t)h * TDc + k];
  float num = da + na * (ab[h] + mod[h]) + dtv + nt * (tb[h] + mod[Hc + h]);
  pooled[b * Hc + h] = num / fmaxf(na + nt, 1.f);
}

__global__ __launch_bounds__(512) void sm_cond(const float* __restrict__ pooled,
                                               const float* __restrict__ cw,
                                               const float* __restrict__ cb,
                                               float* __restrict__ cond) {
  int b = blockIdx.x, h = threadIdx.x;
  const float* p = pooled + b * Hc;
  float a = 0.f;
  for (int k = 0; k < Hc; ++k) a += p[k] * cw[(size_t)h * Hc + k];
  cond[b * Hc + h] = a + cb[h];
}

__global__ __launch_bounds__(256) void sm_add_cond(float* __restrict__ x,
                                                   bf16_t* __restrict__ x_bf,
                                                   const float* __restrict__ cond) {
  int idx = blockIdx.x * 256 + threadIdx.x;
  if (idx >= Mrows * Hc) return;
  int h = idx & (Hc - 1);
  int b = idx >> 17;            // / (L*H) = / 131072
  float v = x[idx] + cond[b * Hc + h];
  x[idx] = v;
  x_bf[idx] = f2bf(v);
}

// ---------------------------------------------------------------------------
// Depthwise causal conv (K=4) + SiLU; reads xin = xz[:, :, 0:DI]
// ---------------------------------------------------------------------------
__global__ __launch_bounds__(256) void sm_conv_silu(float* __restrict__ xc,
                                                    bf16_t* __restrict__ xc_bf,
                                                    const float* __restrict__ xz,
                                                    const float* __restrict__ cw,
                                                    const float* __restrict__ cb) {
  int idx = blockIdx.x * 256 + threadIdx.x;     // (b*L+l)*DI + d
  if (idx >= Mrows * DIc) return;
  int d  = idx & (DIc - 1);
  int bl = idx >> 10;
  int l  = bl & (Lc - 1);
  int b  = bl >> 8;
  float acc = cb[d];
#pragma unroll
  for (int k = 0; k < DCc; ++k) {
    int l2 = l + k - (DCc - 1);
    if (l2 >= 0)
      acc += xz[((size_t)(b * Lc + l2)) * (2 * DIc) + d] * cw[d * DCc + k];
  }
  float s = acc / (1.f + __expf(-acc));
  xc[idx] = s;
  xc_bf[idx] = f2bf(s);
}

// ---------------------------------------------------------------------------
// dt = softplus(dt_raw + dt_b)
// ---------------------------------------------------------------------------
__global__ __launch_bounds__(256) void sm_softplus_bias(float* __restrict__ dt,
                                                        const float* __restrict__ db,
                                                        int n) {
  int i = blockIdx.x * 256 + threadIdx.x;
  if (i >= n) return;
  float v = dt[i] + db[i & (DIc - 1)];
  dt[i] = (v > 20.f) ? v : log1pf(__expf(v));
}

// ---------------------------------------------------------------------------
// Selective scan, fused with D-skip and SiLU(z) gate. One thread per channel,
// h[16] state in registers, B/C tiles staged in 32KB LDS.
// ---------------------------------------------------------------------------
__global__ __launch_bounds__(256) void sm_scan(float* __restrict__ y,
                                               bf16_t* __restrict__ y_bf,
                                               const float* __restrict__ dtv,
                                               const float* __restrict__ dbl,
                                               const float* __restrict__ xc,
                                               const float* __restrict__ xz,
                                               const float* __restrict__ Alog,
                                               const float* __restrict__ Dp) {
  __shared__ float BC[Lc * 2 * DSc];           // 256 * 32 floats = 32 KB
  const int b = blockIdx.x >> 2;
  const int d = ((blockIdx.x & 3) << 8) + threadIdx.x;

  for (int i = threadIdx.x; i < Lc * 2 * DSc; i += 256) {
    int t = i >> 5, j = i & 31;
    BC[i] = dbl[((size_t)(b * Lc + t)) * (DTRc + 2 * DSc) + DTRc + j];
  }
  __syncthreads();

  float Av[DSc], hst[DSc];
#pragma unroll
  for (int s = 0; s < DSc; ++s) {
    Av[s] = -__expf(Alog[d * DSc + s]);
    hst[s] = 0.f;
  }
  const float Dval = Dp[d];

  for (int t = 0; t < Lc; ++t) {
    const size_t base = (size_t)(b * Lc + t);
    const float dt_ = dtv[base * DIc + d];
    const float xc_ = xc[base * DIc + d];
    const float* bc = &BC[t * 2 * DSc];
    float accv = 0.f;
#pragma unroll
    for (int s = 0; s < DSc; ++s) {
      float dA = __expf(dt_ * Av[s]);
      hst[s] = hst[s] * dA + dt_ * bc[s] * xc_;
      accv += hst[s] * bc[DSc + s];
    }
    const float zv = xz[base * (2 * DIc) + DIc + d];
    const float sil = zv / (1.f + __expf(-zv));
    const float yv = (accv + Dval * xc_) * sil;
    y[base * DIc + d] = yv;
    y_bf[base * DIc + d] = f2bf(yv);
  }
}

// ---------------------------------------------------------------------------
// x = LayerNorm(x + yproj) * g + b  (per token row, H=512)
// ---------------------------------------------------------------------------
__global__ __launch_bounds__(256) void sm_residual_ln(float* __restrict__ x,
                                                      bf16_t* __restrict__ x_bf,
                                                      const float* __restrict__ yproj,
                                                      const float* __restrict__ g,
                                                      const float* __restrict__ bta) {
  __shared__ float ssum[256], ssq[256];
  const int row = blockIdx.x;
  const int tid = threadIdx.x;
  const size_t o = (size_t)row * Hc;
  float v0 = x[o + tid]       + yproj[o + tid];
  float v1 = x[o + 256 + tid] + yproj[o + 256 + tid];
  ssum[tid] = v0 + v1;
  ssq[tid]  = v0 * v0 + v1 * v1;
  __syncthreads();
  for (int off = 128; off > 0; off >>= 1) {
    if (tid < off) { ssum[tid] += ssum[tid + off]; ssq[tid] += ssq[tid + off]; }
    __syncthreads();
  }
  const float mu  = ssum[0] * (1.f / Hc);
  const float var = ssq[0] * (1.f / Hc) - mu * mu;
  const float rs  = rsqrtf(var + 1e-5f);
  float o0 = (v0 - mu) * rs * g[tid] + bta[tid];
  float o1 = (v1 - mu) * rs * g[256 + tid] + bta[256 + tid];
  x[o + tid] = o0;          x[o + 256 + tid] = o1;
  x_bf[o + tid] = f2bf(o0); x_bf[o + 256 + tid] = f2bf(o1);
}

// ---------------------------------------------------------------------------
// Host-side orchestration
// ---------------------------------------------------------------------------
extern "C" void kernel_launch(void* const* d_in, const int* in_sizes, int n_in,
                              void* d_out, int out_size, void* d_ws, size_t ws_size,
                              hipStream_t stream) {
  (void)in_sizes; (void)n_in; (void)out_size; (void)ws_size;

  const int*           input_ids  = (const int*)d_in[0];
  const float*         audio_f    = (const float*)d_in[1];
  const float*         text_f     = (const float*)d_in[2];
  const unsigned char* audio_mask = (const unsigned char*)d_in[3];
  const unsigned char* text_mask  = (const unsigned char*)d_in[4];
  const float*         tok_emb    = (const float*)d_in[5];
  const float*         pos_emb    = (const float*)d_in[6];
  const float*         audio_w    = (const float*)d_in[7];
  const float*         audio_b    = (const float*)d_in[8];
  const float*         text_w     = (const float*)d_in[9];
  const float*         text_b     = (const float*)d_in[10];
  const float*         mod_emb    = (const float*)d_in[11];
  const float*         cond_w     = (const float*)d_in[12];
  const float*         cond_b     = (const float*)d_in[13];
  const float*         in_proj_w  = (const float*)d_in[14];
  const float*         conv_w     = (const float*)d_in[15];
  const float*         conv_b     = (const float*)d_in[16];
  const float*         x_proj_w   = (const float*)d_in[17];
  const float*         dt_w       = (const float*)d_in[18];
  const float*         dt_b       = (const float*)d_in[19];
  const float*         A_log      = (const float*)d_in[20];
  const float*         D_skip     = (const float*)d_in[21];
  const float*         out_w      = (const float*)d_in[22];
  const float*         ln_g       = (const float*)d_in[23];
  const float*         ln_b       = (const float*)d_in[24];
  const float*         head_w     = (const float*)d_in[25];
  const float*         head_b     = (const float*)d_in[26];
  float* out = (float*)d_out;

  // ---- workspace carve-out ----
  char*  base = (char*)d_ws;
  size_t off  = 0;
  auto alloc = [&](size_t bytes) -> void* {
    void* p = base + off;
    off += (bytes + 255) & ~(size_t)255;
    return p;
  };
  float*  x      = (float*) alloc((size_t)Mrows * Hc * 4);
  bf16_t* x_bf   = (bf16_t*)alloc((size_t)Mrows * Hc * 2);
  float*  xz     = (float*) alloc((size_t)Mrows * 2 * DIc * 4);
  float*  xc     = (float*) alloc((size_t)Mrows * DIc * 4);
  bf16_t* xc_bf  = (bf16_t*)alloc((size_t)Mrows * DIc * 2);
  float*  dbl    = (float*) alloc((size_t)Mrows * (DTRc + 2 * DSc) * 4);
  bf16_t* dbl_bf = (bf16_t*)alloc((size_t)Mrows * (DTRc + 2 * DSc) * 2);
  float*  dtb    = (float*) alloc((size_t)Mrows * DIc * 4);
  float*  yb     = (float*) alloc((size_t)Mrows * DIc * 4);
  bf16_t* y_bf   = (bf16_t*)alloc((size_t)Mrows * DIc * 2);
  float*  yproj  = (float*) alloc((size_t)Mrows * Hc * 4);
  bf16_t* wbf    = (bf16_t*)alloc((size_t)Vc * Hc * 2);       // reused weight buffer
  float*  sa     = (float*) alloc((size_t)Bc * ADc * 4);
  float*  st     = (float*) alloc((size_t)Bc * TDc * 4);
  float*  cnts   = (float*) alloc((size_t)2 * Bc * 4);
  float*  pooled = (float*) alloc((size_t)Bc * Hc * 4);
  float*  cond   = (float*) alloc((size_t)Bc * Hc * 4);

  auto cvt = [&](bf16_t* dst, const float* src, int n) {   // n % 4 == 0 at all call sites
    int n4 = n >> 2;
    sm_cvt_f32_bf16_x4<<<(n4 + 255) / 256, 256, 0, stream>>>(dst, src, n4);
  };
  auto gemm = [&](float* C, const bf16_t* A, const bf16_t* Bm, const float* bias,
                  int M, int N, int K, int lda, int ldb, int ldc) {
    dim3 grid((N + GBN - 1) / GBN, (M + GBM - 1) / GBM);
    sm_gemm_bf16_wmma<<<grid, 256, 0, stream>>>(C, A, Bm, bias, M, N, K, lda, ldb, ldc);
  };

  // ---- embeddings + conditioning ----
  sm_embed<<<(Mrows * Hc + 255) / 256, 256, 0, stream>>>(x, input_ids, tok_emb, pos_emb);
  sm_mem_reduce<<<Bc, 256, 0, stream>>>(audio_f, audio_mask, TAc, ADc, sa, cnts);
  sm_mem_reduce<<<Bc, 256, 0, stream>>>(text_f, text_mask, TTc, TDc, st, cnts + Bc);
  sm_pooled<<<Bc, Hc, 0, stream>>>(sa, st, cnts, audio_w, audio_b, text_w, text_b,
                                   mod_emb, pooled);
  sm_cond<<<Bc, Hc, 0, stream>>>(pooled, cond_w, cond_b, cond);
  sm_add_cond<<<(Mrows * Hc + 255) / 256, 256, 0, stream>>>(x, x_bf, cond);

  // ---- Mamba layers ----
  for (int i = 0; i < DEPTHc; ++i) {
    // xz = x @ in_proj_w[i]^T : (2048,2048,K=512)
    cvt(wbf, in_proj_w + (size_t)i * 2 * DIc * Hc, 2 * DIc * Hc);
    gemm(xz, x_bf, wbf, nullptr, Mrows, 2 * DIc, Hc, Hc, Hc, 2 * DIc);

    // xc = silu(causal_conv(xin))
    sm_conv_silu<<<(Mrows * DIc + 255) / 256, 256, 0, stream>>>(
        xc, xc_bf, xz, conv_w + (size_t)i * DIc * DCc, conv_b + (size_t)i * DIc);

    // dbl = xc @ x_proj_w[i]^T : (2048,64,K=1024)
    cvt(wbf, x_proj_w + (size_t)i * (DTRc + 2 * DSc) * DIc, (DTRc + 2 * DSc) * DIc);
    gemm(dbl, xc_bf, wbf, nullptr, Mrows, DTRc + 2 * DSc, DIc, DIc, DIc, DTRc + 2 * DSc);
    cvt(dbl_bf, dbl, Mrows * (DTRc + 2 * DSc));

    // dt = softplus(dbl[:, :32] @ dt_w[i]^T + dt_b[i]) : K padded 32->64 with
    // zero weights so the K%64 GEMM pipeline applies (extra dbl columns * 0).
    sm_cvt_dtw_pad<<<(DIc * 64 + 255) / 256, 256, 0, stream>>>(
        wbf, dt_w + (size_t)i * DIc * DTRc);
    gemm(dtb, dbl_bf, wbf, nullptr, Mrows, DIc, 64, DTRc + 2 * DSc, 64, DIc);
    sm_softplus_bias<<<(Mrows * DIc + 255) / 256, 256, 0, stream>>>(
        dtb, dt_b + (size_t)i * DIc, Mrows * DIc);

    // selective scan + D-skip + silu(z) gate
    sm_scan<<<Bc * (DIc / 256), 256, 0, stream>>>(
        yb, y_bf, dtb, dbl, xc, xz, A_log + (size_t)i * DIc * DSc,
        D_skip + (size_t)i * DIc);

    // yproj = y @ out_w[i]^T : (2048,512,K=1024); then residual + LN
    cvt(wbf, out_w + (size_t)i * Hc * DIc, Hc * DIc);
    gemm(yproj, y_bf, wbf, nullptr, Mrows, Hc, DIc, DIc, DIc, Hc);
    sm_residual_ln<<<Mrows, 256, 0, stream>>>(x, x_bf, yproj,
                                              ln_g + (size_t)i * Hc,
                                              ln_b + (size_t)i * Hc);
  }

  // ---- LM head: out = x @ head_w^T + head_b : (2048, 32000, K=512) ----
  cvt(wbf, head_w, Vc * Hc);
  gemm(out, x_bf, wbf, head_b, Mrows, Vc, Hc, Hc, Hc, Vc);
}
